// Speller_901943132838
// MI455X (gfx1250) — matrix operand order repaired and checked by reading, not compiled, not run on hardware
//
#include <hip/hip_runtime.h>

#define BB 64
#define TT 600
#define HH 512
#define VV 33
#define SS 250
#define NEGE (-1e9f)

typedef __bf16 bf16_t;
typedef __attribute__((ext_vector_type(8)))  bf16_t v8bf;
typedef __attribute__((ext_vector_type(16))) bf16_t v16bf;
typedef __attribute__((ext_vector_type(8)))  float  v8f;

// ---------------- setup kernels ----------------

__global__ void k_cvt_bf16(const float* __restrict__ s, bf16_t* __restrict__ d, long long n) {
    long long i = (long long)blockIdx.x * blockDim.x + threadIdx.x;
    long long stride = (long long)gridDim.x * blockDim.x;
    for (; i < n; i += stride) d[i] = (bf16_t)s[i];
}

__global__ void k_add_bias(const float* __restrict__ a, const float* __restrict__ b,
                           float* __restrict__ o, int n) {
    int i = blockIdx.x * blockDim.x + threadIdx.x;
    if (i < n) o[i] = a[i] + b[i];
}

__global__ void k_init_state(const float* __restrict__ ih, const float* __restrict__ ic,
                             float* __restrict__ h, float* __restrict__ c,
                             bf16_t* __restrict__ hbf) {
    int i = blockIdx.x * blockDim.x + threadIdx.x;  // B*H threads
    if (i < BB * HH) {
        float hv = ih[i & (HH - 1)];
        h[i] = hv;
        c[i] = ic[i & (HH - 1)];
        hbf[i] = (bf16_t)hv;
    }
}

__global__ void k_init_word(int* __restrict__ word) {
    int i = threadIdx.x;
    if (i < BB) word[i] = 0;
}

// ---------------- attention (energy -> softmax -> context + embed gather) ----------------
// one block per batch row b; block = 512 threads
__global__ void k_attn(const bf16_t* __restrict__ lis,        // [B,T,H] bf16
                       const float* __restrict__ hq_all,      // h state layer2 [B,H] f32
                       const int* __restrict__ seq,
                       const float* __restrict__ embw,        // [V,H] f32
                       const int* __restrict__ word,
                       bf16_t* __restrict__ x0bf,             // [B,2H]  (emb ++ ctx)
                       bf16_t* __restrict__ xfcbf,            // [B,2H]  (h2  ++ ctx)
                       float* __restrict__ attn_out,          // [B,S,T] section of d_out
                       int s) {
    __shared__ float sE[TT];
    __shared__ float sH[HH];
    __shared__ float sRed[512];

    const int b = blockIdx.x;
    const int tid = threadIdx.x;
    const int len = seq[b];

    sH[tid] = hq_all[b * HH + tid];
    __syncthreads();

    // energy[t] = dot(listener[b,t,:], h)
    float lmax = -3.4e38f;
    for (int t = tid; t < TT; t += 512) {
        const bf16_t* row = lis + ((size_t)b * TT + t) * HH;
        float acc = 0.f;
        #pragma unroll 8
        for (int k = 0; k < HH; ++k) acc += (float)row[k] * sH[k];
        if (t >= len) acc = NEGE;
        sE[t] = acc;
        lmax = fmaxf(lmax, acc);
    }
    sRed[tid] = lmax;
    __syncthreads();
    for (int w = 256; w > 0; w >>= 1) {
        if (tid < w) sRed[tid] = fmaxf(sRed[tid], sRed[tid + w]);
        __syncthreads();
    }
    const float mx = sRed[0];

    float lsum = 0.f;
    for (int t = tid; t < TT; t += 512) {
        float e = __expf(sE[t] - mx);
        sE[t] = e;
        lsum += e;
    }
    __syncthreads();           // everyone has read sRed[0]
    sRed[tid] = lsum;
    __syncthreads();
    for (int w = 256; w > 0; w >>= 1) {
        if (tid < w) sRed[tid] += sRed[tid + w];
        __syncthreads();
    }
    const float inv = 1.0f / sRed[0];

    for (int t = tid; t < TT; t += 512) {
        float p = sE[t] * inv;
        sE[t] = p;
        attn_out[((size_t)b * SS + s) * TT + t] = p;
    }
    __syncthreads();

    // context[h] = sum_t attn[t] * listener[b,t,h]  (coalesced over h)
    {
        const bf16_t* base = lis + (size_t)b * TT * HH + tid;
        float acc = 0.f;
        #pragma unroll 4
        for (int t = 0; t < TT; ++t) acc += sE[t] * (float)base[(size_t)t * HH];
        bf16_t cb = (bf16_t)acc;
        x0bf[(size_t)b * 1024 + HH + tid]  = cb;
        xfcbf[(size_t)b * 1024 + HH + tid] = cb;
        // embedding gather (previous word)
        x0bf[(size_t)b * 1024 + tid] = (bf16_t)embw[(size_t)word[b] * HH + tid];
    }
}

// ---------------- WMMA GEMM:  C[64,N] = A1@W1^T (+ A2@W2^T) + bias, optional leaky ----------------

__device__ inline v8f wmma_pass(v8f acc, const bf16_t* __restrict__ A, const bf16_t* __restrict__ W,
                                int K, int mtile, int ntile, int lm, int hi) {
    const bf16_t* arow = A + (size_t)(mtile + lm) * K + hi * 8;
    const bf16_t* brow = W + (size_t)(ntile + lm) * K + hi * 8;
    for (int k0 = 0; k0 < K; k0 += 32) {
        v8bf alo = *(const v8bf*)(arow + k0);
        v8bf ahi = *(const v8bf*)(arow + k0 + 16);
        v8bf blo = *(const v8bf*)(brow + k0);
        v8bf bhi = *(const v8bf*)(brow + k0 + 16);
        v16bf a = __builtin_shufflevector(alo, ahi, 0,1,2,3,4,5,6,7,8,9,10,11,12,13,14,15);
        v16bf bmat = __builtin_shufflevector(blo, bhi, 0,1,2,3,4,5,6,7,8,9,10,11,12,13,14,15);
        acc = __builtin_amdgcn_wmma_f32_16x16x32_bf16(false, a, false, bmat,
                                                      (short)0, acc, false, false);
    }
    return acc;
}

// block = 256 threads = 8 waves; one wave per 16x16 C tile; M fixed = 64
__global__ void k_gemm(const bf16_t* __restrict__ A1, const bf16_t* __restrict__ W1, int K1,
                       const bf16_t* __restrict__ A2, const bf16_t* __restrict__ W2, int K2,
                       const float* __restrict__ bias, float* __restrict__ Cout,
                       int N, int leaky) {
    const int wave = (blockIdx.x * blockDim.x + threadIdx.x) >> 5;
    const int ntN = N >> 4;
    const int mtile = (wave / ntN) << 4;
    const int ntile = (wave % ntN) << 4;
    const int lane = threadIdx.x & 31;
    const int lm = lane & 15;
    const int hi = lane >> 4;

    v8f acc = {};
    acc = wmma_pass(acc, A1, W1, K1, mtile, ntile, lm, hi);
    if (A2) acc = wmma_pass(acc, A2, W2, K2, mtile, ntile, lm, hi);

    const int col = ntile + lm;
    const float bv = bias[col];
    const int rbase = mtile + hi * 8;
    #pragma unroll
    for (int r = 0; r < 8; ++r) {
        float v = acc[r] + bv;
        if (leaky) v = (v >= 0.f) ? v : 0.2f * v;
        Cout[(size_t)(rbase + r) * N + col] = v;
    }
}

// ---------------- LSTM elementwise gate update ----------------

__device__ inline float sigf(float x) { return 1.0f / (1.0f + __expf(-x)); }

__global__ void k_lstm_elem(const float* __restrict__ gates,   // [B,4H]  (biases already added)
                            float* __restrict__ h, float* __restrict__ c,
                            bf16_t* __restrict__ hbf, bf16_t* __restrict__ xdst) {
    int idx = blockIdx.x * blockDim.x + threadIdx.x;   // B*H
    if (idx >= BB * HH) return;
    int b  = idx >> 9;
    int hh = idx & (HH - 1);
    const float* g = gates + (size_t)b * (4 * HH);
    float ig = sigf(g[hh]);
    float fg = sigf(g[HH + hh]);
    float gg = tanhf(g[2 * HH + hh]);
    float og = sigf(g[3 * HH + hh]);
    float c2 = fg * c[idx] + ig * gg;
    float h2 = og * tanhf(c2);
    c[idx] = c2;
    h[idx] = h2;
    hbf[idx] = (bf16_t)h2;
    if (xdst) xdst[(size_t)b * 1024 + hh] = (bf16_t)h2;   // layer2 -> fc input first half
}

// ---------------- tied unembed + argmax ----------------
// grid = B, block = 64 (V=33 active)
__global__ void k_unembed(const float* __restrict__ z, const float* __restrict__ embw,
                          const float* __restrict__ ub, float* __restrict__ preds,
                          int* __restrict__ word, int s) {
    __shared__ float sp[VV];
    const int b = blockIdx.x;
    const int v = threadIdx.x;
    if (v < VV) {
        const float* zr = z + (size_t)b * HH;
        const float* er = embw + (size_t)v * HH;
        float acc = ub[v];
        #pragma unroll 8
        for (int k = 0; k < HH; ++k) acc += zr[k] * er[k];
        preds[((size_t)b * SS + s) * VV + v] = acc;
        sp[v] = acc;
    }
    __syncthreads();
    if (v == 0) {
        int best = 0;
        float bv = sp[0];
        for (int j = 1; j < VV; ++j) if (sp[j] > bv) { bv = sp[j]; best = j; }
        word[b] = best;
    }
}

// ---------------- host orchestration ----------------

extern "C" void kernel_launch(void* const* d_in, const int* in_sizes, int n_in,
                              void* d_out, int out_size, void* d_ws, size_t ws_size,
                              hipStream_t stream) {
    (void)in_sizes; (void)n_in; (void)out_size; (void)ws_size;

    const float* lis   = (const float*)d_in[0];    // [B,T,H]
    const int*   seq   = (const int*)d_in[1];      // [B]
    const float* embw  = (const float*)d_in[3];    // [V,H]
    const float* unb   = (const float*)d_in[4];    // [V]
    const float* fcw   = (const float*)d_in[5];    // [H,2H]
    const float* fcb   = (const float*)d_in[6];    // [H]
    const float* w_ih[3] = {(const float*)d_in[7],  (const float*)d_in[13], (const float*)d_in[19]};
    const float* w_hh[3] = {(const float*)d_in[8],  (const float*)d_in[14], (const float*)d_in[20]};
    const float* b_ih[3] = {(const float*)d_in[9],  (const float*)d_in[15], (const float*)d_in[21]};
    const float* b_hh[3] = {(const float*)d_in[10], (const float*)d_in[16], (const float*)d_in[22]};
    const float* i_h[3]  = {(const float*)d_in[11], (const float*)d_in[17], (const float*)d_in[23]};
    const float* i_c[3]  = {(const float*)d_in[12], (const float*)d_in[18], (const float*)d_in[24]};

    float* preds = (float*)d_out;                                  // [B,S,V]
    float* attns = (float*)d_out + (size_t)BB * SS * VV;           // [B,S,T]

    // ---- workspace layout (256B aligned slices) ----
    char* wbase = (char*)d_ws;
    size_t off = 0;
    auto take = [&](size_t bytes) -> char* {
        char* p = wbase + off;
        off = (off + bytes + 255) & ~(size_t)255;
        return p;
    };
    bf16_t* lis_bf  = (bf16_t*)take((size_t)BB * TT * HH * 2);
    bf16_t* wihbf[3], *whhbf[3];
    wihbf[0] = (bf16_t*)take((size_t)4 * HH * 2 * HH * 2);   // [2048,1024]
    whhbf[0] = (bf16_t*)take((size_t)4 * HH * HH * 2);       // [2048,512]
    wihbf[1] = (bf16_t*)take((size_t)4 * HH * HH * 2);
    whhbf[1] = (bf16_t*)take((size_t)4 * HH * HH * 2);
    wihbf[2] = (bf16_t*)take((size_t)4 * HH * HH * 2);
    whhbf[2] = (bf16_t*)take((size_t)4 * HH * HH * 2);
    bf16_t* fcwbf = (bf16_t*)take((size_t)HH * 2 * HH * 2);  // [512,1024]
    float*  bsum  = (float*)take((size_t)3 * 4 * HH * 4);    // b_ih + b_hh per layer
    float*  hst   = (float*)take((size_t)3 * BB * HH * 4);
    float*  cst   = (float*)take((size_t)3 * BB * HH * 4);
    bf16_t* hbf   = (bf16_t*)take((size_t)3 * BB * HH * 2);
    bf16_t* x0bf  = (bf16_t*)take((size_t)BB * 2 * HH * 2);  // emb ++ ctx
    bf16_t* xfcbf = (bf16_t*)take((size_t)BB * 2 * HH * 2);  // h2  ++ ctx
    float*  gates = (float*)take((size_t)BB * 4 * HH * 4);
    float*  zbuf  = (float*)take((size_t)BB * HH * 4);
    int*    word  = (int*)take(256);

    // ---- one-time (per launch) conversions & init ----
    auto cvt = [&](const float* s, bf16_t* d, long long n) {
        int blocks = (int)((n + 255) / 256);
        k_cvt_bf16<<<blocks, 256, 0, stream>>>(s, d, n);
    };
    cvt(lis, lis_bf, (long long)BB * TT * HH);
    cvt(w_ih[0], wihbf[0], (long long)4 * HH * 2 * HH);
    cvt(w_hh[0], whhbf[0], (long long)4 * HH * HH);
    cvt(w_ih[1], wihbf[1], (long long)4 * HH * HH);
    cvt(w_hh[1], whhbf[1], (long long)4 * HH * HH);
    cvt(w_ih[2], wihbf[2], (long long)4 * HH * HH);
    cvt(w_hh[2], whhbf[2], (long long)4 * HH * HH);
    cvt(fcw, fcwbf, (long long)HH * 2 * HH);

    for (int l = 0; l < 3; ++l) {
        k_add_bias<<<(4 * HH + 255) / 256, 256, 0, stream>>>(b_ih[l], b_hh[l],
                                                             bsum + (size_t)l * 4 * HH, 4 * HH);
        k_init_state<<<(BB * HH + 255) / 256, 256, 0, stream>>>(
            i_h[l], i_c[l], hst + (size_t)l * BB * HH, cst + (size_t)l * BB * HH,
            hbf + (size_t)l * BB * HH);
    }
    k_init_word<<<1, 64, 0, stream>>>(word);

    // ---- 250 sequential decode steps ----
    const int gatesBlocks = ((4 * HH / 16) * (BB / 16)) / 8;   // 512 waves -> 64 blocks
    const int fcBlocks    = ((HH / 16) * (BB / 16)) / 8;       // 128 waves -> 16 blocks

    for (int s = 0; s < SS; ++s) {
        // attention with layer-2 hidden as query; writes attn output + ctx halves + emb
        k_attn<<<BB, 512, 0, stream>>>(lis_bf, hst + (size_t)2 * BB * HH, seq, embw, word,
                                       x0bf, xfcbf, attns, s);

        // layer 0: x = [emb,ctx] (K=1024), h-path K=512
        k_gemm<<<gatesBlocks, 256, 0, stream>>>(x0bf, wihbf[0], 2 * HH,
                                                hbf, whhbf[0], HH,
                                                bsum, gates, 4 * HH, 0);
        k_lstm_elem<<<(BB * HH + 255) / 256, 256, 0, stream>>>(
            gates, hst, cst, hbf, (bf16_t*)nullptr);

        // layer 1
        k_gemm<<<gatesBlocks, 256, 0, stream>>>(hbf, wihbf[1], HH,
                                                hbf + (size_t)BB * HH, whhbf[1], HH,
                                                bsum + 4 * HH, gates, 4 * HH, 0);
        k_lstm_elem<<<(BB * HH + 255) / 256, 256, 0, stream>>>(
            gates, hst + (size_t)BB * HH, cst + (size_t)BB * HH,
            hbf + (size_t)BB * HH, (bf16_t*)nullptr);

        // layer 2 (new h2 also feeds fc input first half)
        k_gemm<<<gatesBlocks, 256, 0, stream>>>(hbf + (size_t)BB * HH, wihbf[2], HH,
                                                hbf + (size_t)2 * BB * HH, whhbf[2], HH,
                                                bsum + 8 * HH, gates, 4 * HH, 0);
        k_lstm_elem<<<(BB * HH + 255) / 256, 256, 0, stream>>>(
            gates, hst + (size_t)2 * BB * HH, cst + (size_t)2 * BB * HH,
            hbf + (size_t)2 * BB * HH, xfcbf);

        // fc: [h2,ctx] @ fc_w^T + fc_b, LeakyReLU(0.2)
        k_gemm<<<fcBlocks, 256, 0, stream>>>(xfcbf, fcwbf, 2 * HH,
                                             (const bf16_t*)nullptr, (const bf16_t*)nullptr, 0,
                                             fcb, zbuf, HH, 1);

        // tied unembed + argmax -> next word
        k_unembed<<<BB, 64, 0, stream>>>(zbuf, embw, unb, preds, word, s);
    }
}
